// GCN_60052232732714
// MI455X (gfx1250) — matrix operand order repaired and checked by reading, not compiled, not run on hardware
//
#include <hip/hip_runtime.h>
#include <hip/hip_bf16.h>

typedef __attribute__((ext_vector_type(16))) __bf16 v16bf;
typedef __attribute__((ext_vector_type(8)))  float  v8f;

#define GCN_N 100000

// ---------------- degree / norm ----------------
__global__ void init_deg_kernel(float* deg, int n) {
  int i = blockIdx.x * 256 + threadIdx.x;
  if (i < n) deg[i] = 1.0f;                       // self-loop contributes 1
}

__global__ void count_deg_kernel(const long long* __restrict__ dst, float* deg, int E) {
  int e = blockIdx.x * 256 + threadIdx.x;
  if (e < E) atomicAdd(&deg[(int)dst[e]], 1.0f);
}

__global__ void dinv_kernel(float* deg, int n) {
  int i = blockIdx.x * 256 + threadIdx.x;
  if (i < n) deg[i] = rsqrtf(deg[i]);             // deg >= 1 always (self-loop)
}

// ---------------- bf16 hi/lo split WMMA GEMM: C[M x NCOLS] = A[M x 128] @ B[128 x NCOLS] ----------------
// One wave computes one 16x16 tile. Block = NCOLS/16 waves covering 16 rows x NCOLS cols.
template <int NCOLS, bool RELU_A>
__global__ __launch_bounds__(256) void gemm_kernel(const float* __restrict__ A,
                                                   const float* __restrict__ B,
                                                   float* __restrict__ C) {
  constexpr int K = 128;
  const int wave = threadIdx.x >> 5;
  const int lane = threadIdx.x & 31;
  const int half = lane >> 4;        // K-half selector (ISA A/B fragment layout)
  const int l15  = lane & 15;
  const int row0 = blockIdx.x << 4;
  const int col0 = wave << 4;

  const float* arow = A + (long)(row0 + l15) * K;
  const int    n    = col0 + l15;

  v8f acc = {0.f, 0.f, 0.f, 0.f, 0.f, 0.f, 0.f, 0.f};

#pragma unroll
  for (int k0 = 0; k0 < K; k0 += 32) {
    v16bf ahi, alo, bhi, blo;
    // A fragment: 16-bit A 16x32 layout (lanes 0-15: K pairs {0,1..6,7,16..23}; lanes 16-31: +8)
#pragma unroll
    for (int v = 0; v < 8; ++v) {
      int k = k0 + ((v & 4) << 2) + (half << 3) + ((v & 3) << 1);
      float f0 = arow[k];
      float f1 = arow[k + 1];
      if (RELU_A) { f0 = fmaxf(f0, 0.f); f1 = fmaxf(f1, 0.f); }
      __bf16 h0 = (__bf16)f0; ahi[2 * v]     = h0; alo[2 * v]     = (__bf16)(f0 - (float)h0);
      __bf16 h1 = (__bf16)f1; ahi[2 * v + 1] = h1; alo[2 * v + 1] = (__bf16)(f1 - (float)h1);
    }
    // B fragment: 32x16, same K order as A (lanes 0-15: K {0..7,16..23}; lanes 16-31: +8)
#pragma unroll
    for (int v = 0; v < 8; ++v) {
      int k = k0 + ((v & 4) << 2) + (half << 3) + ((v & 3) << 1);
      float f0 = B[(long)k * NCOLS + n];
      float f1 = B[(long)(k + 1) * NCOLS + n];
      __bf16 h0 = (__bf16)f0; bhi[2 * v]     = h0; blo[2 * v]     = (__bf16)(f0 - (float)h0);
      __bf16 h1 = (__bf16)f1; bhi[2 * v + 1] = h1; blo[2 * v + 1] = (__bf16)(f1 - (float)h1);
    }
    // fp32-class accuracy on the bf16 pipe: Ahi*Bhi + Alo*Bhi + Ahi*Blo
    acc = __builtin_amdgcn_wmma_f32_16x16x32_bf16(false, ahi, false, bhi, (short)0, acc, false, false);
    acc = __builtin_amdgcn_wmma_f32_16x16x32_bf16(false, alo, false, bhi, (short)0, acc, false, false);
    acc = __builtin_amdgcn_wmma_f32_16x16x32_bf16(false, ahi, false, blo, (short)0, acc, false, false);
  }

  // C/D layout: VGPR r -> row r (lanes 0-15) / row 8+r (lanes 16-31), col = lane%16
  float* cptr = C + (long)row0 * NCOLS + col0 + l15;
#pragma unroll
  for (int r = 0; r < 8; ++r) {
    cptr[(long)(r + (half << 3)) * NCOLS] = acc[r];
  }
}

// ---------------- self-loop + bias init: out[i,:] = dinv[i]^2 * h[i,:] + b ----------------
template <int D>
__global__ void selfloop_bias_kernel(const float* __restrict__ h, const float* __restrict__ dinv,
                                     const float* __restrict__ b, float* __restrict__ out) {
  constexpr int P = D / 4;
  long t = (long)blockIdx.x * 256 + threadIdx.x;
  long total = (long)GCN_N * P;
  if (t >= total) return;
  int node = (int)(t / P);
  int j    = (int)(t % P);
  float di = dinv[node];
  float w  = di * di;
  float4 v  = ((const float4*)h)[t];
  float4 bb = ((const float4*)b)[j];
  float4 r;
  r.x = fmaf(w, v.x, bb.x);
  r.y = fmaf(w, v.y, bb.y);
  r.z = fmaf(w, v.z, bb.z);
  r.w = fmaf(w, v.w, bb.w);
  ((float4*)out)[t] = r;
}

// ---------------- edge scatter: out[dst] += dinv[src]*dinv[dst] * h[src] ----------------
template <int D>
__global__ void scatter_kernel(const float* __restrict__ h, const long long* __restrict__ src,
                               const long long* __restrict__ dst, const float* __restrict__ dinv,
                               float* __restrict__ out, int E) {
  constexpr int L = D / 4;  // lanes per edge (float4 per lane)
  long t = (long)blockIdx.x * 256 + threadIdx.x;
  long e = t / L;
  int  j = (int)(t % L);
  if (e >= E) return;
  int s = (int)src[e];
  int d = (int)dst[e];
  float nrm = dinv[s] * dinv[d];
  float4 v = ((const float4*)(h + (long)s * D))[j];
  float* o = out + (long)d * D + j * 4;
  atomicAdd(o + 0, nrm * v.x);
  atomicAdd(o + 1, nrm * v.y);
  atomicAdd(o + 2, nrm * v.z);
  atomicAdd(o + 3, nrm * v.w);
}

// ---------------- launcher ----------------
extern "C" void kernel_launch(void* const* d_in, const int* in_sizes, int n_in,
                              void* d_out, int out_size, void* d_ws, size_t ws_size,
                              hipStream_t stream) {
  (void)n_in; (void)out_size; (void)ws_size;
  const float*     x   = (const float*)d_in[0];
  const long long* ei  = (const long long*)d_in[1];   // int64 [2, E]
  const float*     W1  = (const float*)d_in[2];
  const float*     b1  = (const float*)d_in[3];
  const float*     W2  = (const float*)d_in[4];
  const float*     b2  = (const float*)d_in[5];
  float*           out = (float*)d_out;

  const int N = GCN_N;
  const int E = in_sizes[1] / 2;
  const long long* src = ei;
  const long long* dst = ei + E;

  // workspace: dinv [N] | hbuf [N*128] (reused for h2) | aggbuf [N*128]
  float* ws     = (float*)d_ws;
  float* dinv   = ws;
  float* hbuf   = ws + 100096;
  float* aggbuf = hbuf + (long)N * 128;

  // degrees -> dinv (self-loops folded in as +1)
  init_deg_kernel<<<(N + 255) / 256, 256, 0, stream>>>(dinv, N);
  count_deg_kernel<<<(E + 255) / 256, 256, 0, stream>>>(dst, dinv, E);
  dinv_kernel<<<(N + 255) / 256, 256, 0, stream>>>(dinv, N);

  // layer 1: h = x @ W1 (WMMA), agg = dinv^2*h + b1, agg += edge messages
  gemm_kernel<128, false><<<N / 16, 256, 0, stream>>>(x, W1, hbuf);
  {
    long t = (long)N * 32;
    selfloop_bias_kernel<128><<<(int)((t + 255) / 256), 256, 0, stream>>>(hbuf, dinv, b1, aggbuf);
  }
  {
    long t = (long)E * 32;
    scatter_kernel<128><<<(int)((t + 255) / 256), 256, 0, stream>>>(hbuf, src, dst, dinv, aggbuf, E);
  }

  // layer 2: h2 = relu(agg) @ W2 (ReLU fused into A-load), out = dinv^2*h2 + b2, out += messages
  gemm_kernel<64, true><<<N / 16, 128, 0, stream>>>(aggbuf, W2, hbuf);
  {
    long t = (long)N * 16;
    selfloop_bias_kernel<64><<<(int)((t + 255) / 256), 256, 0, stream>>>(hbuf, dinv, b2, out);
  }
  {
    long t = (long)E * 16;
    scatter_kernel<64><<<(int)((t + 255) / 256), 256, 0, stream>>>(hbuf, src, dst, dinv, out, E);
  }
}
